// Self_Attention_73753178407400
// MI455X (gfx1250) — compile-verified, matrix operand order checked by reading, not run on hardware
//
#include <hip/hip_runtime.h>
#include <hip/hip_bf16.h>
#include <math.h>

typedef __attribute__((ext_vector_type(16))) _Float16 v16h;
typedef __attribute__((ext_vector_type(8)))  _Float16 v8h;
typedef __attribute__((ext_vector_type(8)))  float    v8f;

#define BATCH 4
#define CCH   256
#define NTOK  4096
#define DQK   32
#define WAVES 4          // waves per workgroup
#define DOUT  320        // 32 (q) + 32 (k) + 256 (v) projection rows

// ---------------------------------------------------------------------------
// Fragment loader: 16 rows x 32 k-halves, two contiguous 16B vector loads.
//   lane 0-15 : row = lane,    halves 0-7 -> k 0..7,  8-15 -> k 16..23
//   lane16-31 : row = lane-16, halves 0-7 -> k 8..15, 8-15 -> k 24..31
// ---------------------------------------------------------------------------
__device__ __forceinline__ v16h load_frag(const _Float16* p, int ld,
                                          int r15, int kb)
{
    const _Float16* row = p + (size_t)r15 * ld;
    v8h lo = *(const v8h*)(row + kb);
    v8h hi = *(const v8h*)(row + 16 + kb);
    return __builtin_shufflevector(lo, hi, 0,1,2,3,4,5,6,7,
                                          8,9,10,11,12,13,14,15);
}

// Async global->LDS 16B copy (CDNA5 ASYNCcnt path, no VGPR round-trip).
__device__ __forceinline__ void async_copy16(uint32_t lds_addr, uint64_t gaddr)
{
    asm volatile("global_load_async_to_lds_b128 %0, %1, off"
                 :: "v"(lds_addr), "v"(gaddr) : "memory");
}
__device__ __forceinline__ void wait_async0()
{
    asm volatile("s_wait_asynccnt 0" ::: "memory");
}
// Compiler-only ordering for intra-wave LDS RAW (HW keeps DS ops in order).
__device__ __forceinline__ void lds_order() { asm volatile("" ::: "memory"); }

// ---------------------------------------------------------------------------
// Kernel 0: pack [wq;wk;wv] -> f16 Wh[320][256], biases -> bf[320].
// ---------------------------------------------------------------------------
__global__ void __launch_bounds__(256)
pack_weights_kernel(const float* __restrict__ wq, const float* __restrict__ bq,
                    const float* __restrict__ wk, const float* __restrict__ bk,
                    const float* __restrict__ wv, const float* __restrict__ bv,
                    _Float16* __restrict__ Wh, float* __restrict__ bf)
{
    const int r = blockIdx.x;           // 0..319
    const int t = threadIdx.x;          // 0..255
    const float* src = (r < 32) ? (wq + (size_t)r * CCH)
                     : (r < 64) ? (wk + (size_t)(r - 32) * CCH)
                                : (wv + (size_t)(r - 64) * CCH);
    Wh[(size_t)r * CCH + t] = (_Float16)src[t];
    if (t == 0)
        bf[r] = (r < 32) ? bq[r] : (r < 64) ? bk[r - 32] : bv[r - 64];
}

// ---------------------------------------------------------------------------
// Kernel 1: transpose + convert x[B,C,N] f32 -> Xt[B,N,C] f16 (LDS tiled).
// ---------------------------------------------------------------------------
__global__ void __launch_bounds__(256)
xpose_kernel(const float* __restrict__ x, _Float16* __restrict__ Xt)
{
    __shared__ float tile[32][33];
    const int b  = blockIdx.z;
    const int n0 = blockIdx.x * 32;
    const int c0 = blockIdx.y * 32;
    const int tx = threadIdx.x;         // 0..31
    const int ty = threadIdx.y;         // 0..7
#pragma unroll
    for (int i = 0; i < 4; ++i)
        tile[ty + i * 8][tx] =
            x[((size_t)b * CCH + (c0 + ty + i * 8)) * NTOK + n0 + tx];
    __syncthreads();
#pragma unroll
    for (int i = 0; i < 4; ++i)
        Xt[((size_t)b * NTOK + n0 + ty + i * 8) * CCH + c0 + tx] =
            (_Float16)tile[tx][ty + i * 8];
}

// ---------------------------------------------------------------------------
// Kernel 2: WMMA projection.  D[320, N] = Wh[320,256] x Xt^T + bias.
// Wave owns 16 tokens; 8 hoisted X B-fragments; 20 row-tiles x 8 k-steps.
// Store loops split by destination so branches resolve at compile time.
// ---------------------------------------------------------------------------
__global__ void __launch_bounds__(WAVES * 32)
proj_wmma_kernel(const _Float16* __restrict__ Wh, const float* __restrict__ bf,
                 const _Float16* __restrict__ Xt,
                 _Float16* __restrict__ Qh, _Float16* __restrict__ Kh,
                 _Float16* __restrict__ Vc)
{
    const int tid  = threadIdx.x;
    const int lane = tid & 31;
    const int w    = tid >> 5;
    const int b    = blockIdx.y;
    const int n0   = (blockIdx.x * WAVES + w) * 16;

    const int r15 = lane & 15;
    const int hi  = lane >> 4;
    const int kb  = hi << 3;
    const int n   = n0 + r15;

    const _Float16* Xb = Xt + ((size_t)b * NTOK + n0) * CCH;

    v16h bx[8];
#pragma unroll
    for (int k = 0; k < 8; ++k)
        bx[k] = load_frag(Xb + k * 32, CCH, r15, kb);

    auto gemm_tile = [&](int d0) -> v8f {
        v8f acc = (v8f){0.f,0.f,0.f,0.f,0.f,0.f,0.f,0.f};
#pragma unroll
        for (int k = 0; k < 8; ++k) {
            v16h af = load_frag(Wh + (size_t)(d0 * 16) * CCH + k * 32,
                                CCH, r15, kb);
            acc = __builtin_amdgcn_wmma_f32_16x16x32_f16(
                false, af, false, bx[k], (short)0, acc, false, false);
        }
        return acc;
    };

#pragma unroll
    for (int d0 = 0; d0 < 2; ++d0) {                    // Q rows 0..31
        v8f acc = gemm_tile(d0);
#pragma unroll
        for (int r = 0; r < 8; ++r) {
            const int d = d0 * 16 + r + hi * 8;
            Qh[((size_t)b * NTOK + n) * DQK + d] = (_Float16)(acc[r] + bf[d]);
        }
    }
#pragma unroll
    for (int d0 = 2; d0 < 4; ++d0) {                    // K rows 32..63
        v8f acc = gemm_tile(d0);
#pragma unroll
        for (int r = 0; r < 8; ++r) {
            const int d = d0 * 16 + r + hi * 8;
            Kh[((size_t)b * NTOK + n) * DQK + (d - 32)] = (_Float16)(acc[r] + bf[d]);
        }
    }
#pragma unroll 4
    for (int d0 = 4; d0 < DOUT / 16; ++d0) {            // V rows 64..319
        v8f acc = gemm_tile(d0);
#pragma unroll
        for (int r = 0; r < 8; ++r) {
            const int d = d0 * 16 + r + hi * 8;
            Vc[((size_t)b * CCH + (d - 64)) * NTOK + n] = (_Float16)(acc[r] + bf[d]);
        }
    }
}

// ---------------------------------------------------------------------------
// Kernel 3: flash-attention. 4 waves/block; wave w owns 16 queries; all
// waves share double-buffered async-staged K/V chunks (32 keys) in LDS.
// ---------------------------------------------------------------------------
__global__ void __launch_bounds__(WAVES * 32)
flash_attn_kernel(const _Float16* __restrict__ Qh,
                  const _Float16* __restrict__ Kh,
                  const _Float16* __restrict__ Vc,
                  const float* __restrict__ x,
                  const float* __restrict__ gamma,
                  float* __restrict__ out)
{
    const int tid  = threadIdx.x;            // 0..127
    const int lane = tid & 31;
    const int w    = tid >> 5;               // wave id 0..3
    const int b    = blockIdx.y;
    const int i0   = (blockIdx.x * WAVES + w) * 16;

    const int r15 = lane & 15;
    const int hi  = lane >> 4;               // 0 or 1
    const int kb  = hi << 3;                 // 0 or 8

    __shared__ _Float16 Kbuf[2][32][DQK];      //  4 KB
    __shared__ _Float16 Vbuf[2][CCH][32];      // 32 KB
    __shared__ float    Sbuf[WAVES][16][32];   //  8 KB
    __shared__ _Float16 Pbuf[WAVES][16][32];   //  4 KB
    __shared__ float    rowscale[WAVES][16];
    __shared__ float    rowinv[WAVES][16];

    const _Float16* Qb = Qh + (size_t)b * NTOK * DQK;
    const _Float16* Kb = Kh + (size_t)b * NTOK * DQK;
    const _Float16* Vb = Vc + (size_t)b * CCH * NTOK;

    const v16h aq = load_frag(Qb + (size_t)i0 * DQK, DQK, r15, kb);

    v8f Oacc[16];
#pragma unroll
    for (int t = 0; t < 16; ++t) Oacc[t] = (v8f){0.f,0.f,0.f,0.f,0.f,0.f,0.f,0.f};

    // Row stats: lane pair (L, L^16) both track row r15 (kept identical).
    float mrow = -INFINITY, lrow = 0.0f;

    auto stage = [&](int buf, int j0) {
        {   // K chunk: 128 x 16B granules, one per thread
            const int j = tid >> 2, q = tid & 3;
            async_copy16((uint32_t)(uintptr_t)&Kbuf[buf][j][q * 8],
                         (uint64_t)(uintptr_t)(Kb + (size_t)(j0 + j) * DQK + q * 8));
        }
#pragma unroll
        for (int it = 0; it < 8; ++it) {     // V chunk: 1024 granules
            const int g = tid + it * 128;
            const int c = g >> 2, q = g & 3;
            async_copy16((uint32_t)(uintptr_t)&Vbuf[buf][c][q * 8],
                         (uint64_t)(uintptr_t)(Vb + (size_t)c * NTOK + j0 + q * 8));
        }
    };

    stage(0, 0);                             // prologue

    const int NCH = NTOK / 32;
    for (int jc = 0; jc < NCH; ++jc) {
        const int cur = jc & 1;
        const int nxt = cur ^ 1;

        wait_async0();                       // cur buffer resident
        __syncthreads();                     // all waves done with nxt's old data

        if (jc + 1 < NCH) stage(nxt, (jc + 1) * 32);   // overlap next chunk

        // ---- S = Q K^T : two 16x16 tiles, K-dim = 32 (from LDS) ----
        v16h bk0 = load_frag(&Kbuf[cur][0][0],  DQK, r15, kb);
        v16h bk1 = load_frag(&Kbuf[cur][16][0], DQK, r15, kb);
        v8f s0 = (v8f){0.f,0.f,0.f,0.f,0.f,0.f,0.f,0.f};
        v8f s1 = (v8f){0.f,0.f,0.f,0.f,0.f,0.f,0.f,0.f};
        s0 = __builtin_amdgcn_wmma_f32_16x16x32_f16(false, aq, false, bk0,
                                                    (short)0, s0, false, false);
        s1 = __builtin_amdgcn_wmma_f32_16x16x32_f16(false, aq, false, bk1,
                                                    (short)0, s1, false, false);

        // ---- spill S to per-wave LDS (C-layout: vgpr r -> row r + 8*hi) ----
#pragma unroll
        for (int r = 0; r < 8; ++r) {
            Sbuf[w][r + hi * 8][r15]      = s0[r];
            Sbuf[w][r + hi * 8][16 + r15] = s1[r];
        }
        lds_order();                         // same-wave DS ops are in-order

        // ---- online softmax, all 32 lanes: pair (L, L^16) owns row r15;
        //      each lane handles 16 of the 32 columns, combine via shfl ----
        {
            const int jb = hi * 16;
            float cmax = -INFINITY;
#pragma unroll
            for (int j = 0; j < 16; ++j)
                cmax = fmaxf(cmax, Sbuf[w][r15][jb + j]);
            cmax = fmaxf(cmax, __shfl_xor(cmax, 16, 32));
            const float mnew  = fmaxf(mrow, cmax);
            const float scale = __expf(mrow - mnew);
            float psum = 0.0f;
#pragma unroll
            for (int j = 0; j < 16; ++j) {
                const float p = __expf(Sbuf[w][r15][jb + j] - mnew);
                Pbuf[w][r15][jb + j] = (_Float16)p;
                psum += p;
            }
            psum += __shfl_xor(psum, 16, 32);
            lrow = lrow * scale + psum;      // identical in both pair lanes
            mrow = mnew;
            rowscale[w][r15] = scale;        // pair writes same value
        }
        lds_order();

        // ---- rescale accumulators ----
        float sc[8];
#pragma unroll
        for (int r = 0; r < 8; ++r) sc[r] = rowscale[w][r + hi * 8];
#pragma unroll
        for (int t = 0; t < 16; ++t)
#pragma unroll
            for (int r = 0; r < 8; ++r) Oacc[t][r] *= sc[r];

        // ---- O += P * V : P from LDS as A-frag, V tiles from shared LDS ----
        const v16h pa = load_frag(&Pbuf[w][0][0], 32, r15, kb);
#pragma unroll
        for (int t = 0; t < 16; ++t) {
            v16h bv = load_frag(&Vbuf[cur][t * 16][0], 32, r15, kb);
            Oacc[t] = __builtin_amdgcn_wmma_f32_16x16x32_f16(
                false, pa, false, bv, (short)0, Oacc[t], false, false);
        }
    }

    // ---- finalize: /= row-sum, out = gamma*O + x (vectorized b128 I/O) ----
    rowinv[w][r15] = 1.0f / lrow;            // pair writes same value
    lds_order();
    float inv[8];
#pragma unroll
    for (int r = 0; r < 8; ++r) inv[r] = rowinv[w][r + hi * 8];

    const float g = gamma[0];
#pragma unroll
    for (int t = 0; t < 16; ++t) {
        const int c = t * 16 + r15;
        // rows r=0..7 map to consecutive i = i0 + hi*8 + r
        const size_t base = ((size_t)b * CCH + c) * NTOK + i0 + hi * 8;
        const float4 x0 = *(const float4*)(x + base);
        const float4 x1 = *(const float4*)(x + base + 4);
        float4 o0, o1;
        o0.x = g * (Oacc[t][0] * inv[0]) + x0.x;
        o0.y = g * (Oacc[t][1] * inv[1]) + x0.y;
        o0.z = g * (Oacc[t][2] * inv[2]) + x0.z;
        o0.w = g * (Oacc[t][3] * inv[3]) + x0.w;
        o1.x = g * (Oacc[t][4] * inv[4]) + x1.x;
        o1.y = g * (Oacc[t][5] * inv[5]) + x1.y;
        o1.z = g * (Oacc[t][6] * inv[6]) + x1.z;
        o1.w = g * (Oacc[t][7] * inv[7]) + x1.w;
        *(float4*)(out + base)     = o0;
        *(float4*)(out + base + 4) = o1;
    }
}

// ---------------------------------------------------------------------------
// Launch. Inputs: x, wq, bq, wk, bk, wv, bv, gamma
// Workspace: Qh @0 (1MB) | Kh @1MB | Vc @2MB (8MB) | Xt @10MB (8MB)
//            Wh @18MB (160KB) | bf @18MB+256KB
// ---------------------------------------------------------------------------
extern "C" void kernel_launch(void* const* d_in, const int* in_sizes, int n_in,
                              void* d_out, int out_size, void* d_ws, size_t ws_size,
                              hipStream_t stream)
{
    const float* x     = (const float*)d_in[0];
    const float* wq    = (const float*)d_in[1];
    const float* bq    = (const float*)d_in[2];
    const float* wk    = (const float*)d_in[3];
    const float* bk    = (const float*)d_in[4];
    const float* wv    = (const float*)d_in[5];
    const float* bv    = (const float*)d_in[6];
    const float* gamma = (const float*)d_in[7];
    float* out = (float*)d_out;

    char* ws = (char*)d_ws;
    _Float16* Qh = (_Float16*)(ws);
    _Float16* Kh = (_Float16*)(ws + (size_t)(1 << 20));
    _Float16* Vc = (_Float16*)(ws + (size_t)(2 << 20));
    _Float16* Xt = (_Float16*)(ws + (size_t)(10 << 20));
    _Float16* Wh = (_Float16*)(ws + (size_t)(18 << 20));
    float*    bf = (float*)   (ws + (size_t)(18 << 20) + (256 << 10));

    hipLaunchKernelGGL(pack_weights_kernel, dim3(DOUT), dim3(256), 0, stream,
                       wq, bq, wk, bk, wv, bv, Wh, bf);

    hipLaunchKernelGGL(xpose_kernel, dim3(NTOK / 32, CCH / 32, BATCH),
                       dim3(32, 8), 0, stream, x, Xt);

    hipLaunchKernelGGL(proj_wmma_kernel, dim3(NTOK / (16 * WAVES), BATCH),
                       dim3(WAVES * 32), 0, stream, Wh, bf, Xt, Qh, Kh, Vc);

    hipLaunchKernelGGL(flash_attn_kernel, dim3(NTOK / (16 * WAVES), BATCH),
                       dim3(WAVES * 32), 0, stream, Qh, Kh, Vc, x, gamma, out);
}
